// TransformerBlockQuantum_65481071410429
// MI455X (gfx1250) — matrix operand order repaired and checked by reading, not compiled, not run on hardware
//
#include <hip/hip_runtime.h>
#include <math.h>
#include <stdint.h>

// ---------------------------------------------------------------------------
// Quantum transformer block, MI455X (gfx1250, wave32, WMMA).
//   k1: q features (cos / cumprod)  -> f16 q in ws
//   k2: flash-attention (q==K==V) + W_comb + residual + LN1 -> f32 x1 in ws
//       key tiles double-buffered in LDS via GLOBAL_LOAD_ASYNC_TO_LDS_B128
//   k3: quantum FFN (W1 VALU, W2 via WMMA) + residual + LN2 -> d_out
// Workspace: 2 MB (q f16) + 4 MB (x1 f32) = 6 MB.
// ---------------------------------------------------------------------------

typedef __attribute__((ext_vector_type(16))) _Float16 v16h;
typedef __attribute__((ext_vector_type(8)))  _Float16 h8;
typedef __attribute__((ext_vector_type(8)))  float    v8f;

#define B_    8
#define S_    2048
#define H_    8
#define Dh_   8
#define E_    64
#define FFN_  256
#define NQ_   8
#define ROWS_ (B_ * S_)
#define EPS_  1e-5f
#define SCALE_ 0.35355339059327373f  // 1/sqrt(8)

#if defined(__gfx1250__) && __has_builtin(__builtin_amdgcn_global_load_async_to_lds_b128)
#define QTB_ASYNC_LDS 1
#else
#define QTB_ASYNC_LDS 0
#endif

#if QTB_ASYNC_LDS
// builtin parameter types revealed by clang diagnostics:
//   arg0: int __attribute__((vector_size(16))) __device__ *   (global, AS1)
//   arg1: same pointee in __shared__ (AS3); arg2/arg3: imm ints
typedef int qtb_v4i __attribute__((vector_size(16)));
typedef qtb_v4i __attribute__((address_space(1)))* qtb_as1_v4i_p;
typedef qtb_v4i __attribute__((address_space(3)))* qtb_as3_v4i_p;
#endif

__device__ __forceinline__ float redsum16(float v) {
  v += __shfl_xor(v, 1, 32);
  v += __shfl_xor(v, 2, 32);
  v += __shfl_xor(v, 4, 32);
  v += __shfl_xor(v, 8, 32);
  return v;
}
__device__ __forceinline__ float redmax16(float v) {
  v = fmaxf(v, __shfl_xor(v, 1, 32));
  v = fmaxf(v, __shfl_xor(v, 2, 32));
  v = fmaxf(v, __shfl_xor(v, 4, 32));
  v = fmaxf(v, __shfl_xor(v, 8, 32));
  return v;
}
__device__ __forceinline__ v8f wmma_f32_f16(v16h a, v16h b, v8f c) {
  return __builtin_amdgcn_wmma_f32_16x16x32_f16(false, a, false, b,
                                                (short)0, c, false, false);
}

// async global(16B/lane) -> LDS copy; fallback goes through registers.
__device__ __forceinline__ void stage16_async(const _Float16* gsrc,
                                              _Float16* ldst) {
#if QTB_ASYNC_LDS
  __builtin_amdgcn_global_load_async_to_lds_b128(
      (qtb_as1_v4i_p)(uintptr_t)gsrc,
      (qtb_as3_v4i_p)(uint32_t)(uintptr_t)ldst,
      0, 0);
#else
  *(h8*)ldst = *(const h8*)gsrc;
#endif
}
__device__ __forceinline__ void wait_async_all() {
#if QTB_ASYNC_LDS
#if __has_builtin(__builtin_amdgcn_s_wait_asynccnt)
  __builtin_amdgcn_s_wait_asynccnt(0);
#else
  asm volatile("s_wait_asynccnt 0" ::: "memory");
#endif
#endif
}

// ---------------------------------------------------------------------------
// Kernel 1: quantum 'attention' features.
// q[:,0] = prod_{i=1..7} cos(x_i+th_i); q[:,j>=1] = prod_{i<=j} cos(x_i+th_i)
// ---------------------------------------------------------------------------
__global__ __launch_bounds__(256) void qtb_qfeat(
    const float* __restrict__ x, const float* __restrict__ theta,
    _Float16* __restrict__ q) {
  int tid = blockIdx.x * 256 + threadIdx.x;
  if (tid >= ROWS_ * H_) return;
  const float* xr = x + (long)tid * Dh_;
  float c[8];
#pragma unroll
  for (int i = 0; i < 8; ++i) c[i] = __cosf(xr[i] + theta[i]);
  float cp[8];
  cp[0] = c[0];
#pragma unroll
  for (int i = 1; i < 8; ++i) cp[i] = cp[i - 1] * c[i];
  float z0 = c[1];
#pragma unroll
  for (int i = 2; i < 8; ++i) z0 *= c[i];
  h8 o;
  o[0] = (_Float16)z0;
#pragma unroll
  for (int i = 1; i < 8; ++i) o[i] = (_Float16)cp[i];
  *(h8*)(q + (long)tid * 8) = o;
}

// ---------------------------------------------------------------------------
// Kernel 2: fused flash attention (Q=K=V=q) + W_comb + residual + LN1.
// grid = (S/128, B). 256 threads = 8 wave32; each wave owns 16 rows.
// Key tiles (32 keys x 64 feats) double-buffered in LDS; row-major copy is
// staged asynchronously, transposed copy built from a pipelined register load.
// ---------------------------------------------------------------------------
__global__ __launch_bounds__(256) void qtb_attn(
    const _Float16* __restrict__ q, const float* __restrict__ x,
    const float* __restrict__ Wc, const float* __restrict__ bcomb,
    const float* __restrict__ g1, const float* __restrict__ lnb1,
    float* __restrict__ x1) {
  constexpr int KPAD = 72;   // kv row pad (halves): 144B rows
  constexpr int KTP  = 40;   // ktr row pad
  constexpr int PPAD = 40;   // P tile row pad
  constexpr int OPAD = 68;   // O f32 tile row pad
  constexpr int WPAD = 72;   // W_comb f16 row pad
  constexpr int KVE  = 32 * KPAD;  // elems per kv buffer
  constexpr int KTE  = 64 * KTP;   // elems per ktr buffer
  constexpr int PBE  = 16 * PPAD;  // elems per wave P tile
  constexpr int OBE  = 16 * OPAD;  // elems per wave O tile
  constexpr int WLB  = 64 * WPAD * 2;          // 9216 B
  constexpr int KVB  = 2 * KVE * 2;            // 9216 B
  constexpr int KTB  = 2 * KTE * 2;            // 10240 B
  constexpr int PBB  = 8 * PBE * 2;            // 10240 B
  constexpr int OBB  = 8 * OBE * 4;            // 34816 B (epilogue alias)
  constexpr int UNIB = (KVB + KTB + PBB) > OBB ? (KVB + KTB + PBB) : OBB;
  __shared__ alignas(16) char smem[WLB + UNIB];  // 44032 B

  _Float16* wl   = (_Float16*)smem;
  char*     uni  = smem + WLB;
  _Float16* kvb  = (_Float16*)uni;               // [2][32*KPAD]
  _Float16* ktrb = (_Float16*)(uni + KVB);       // [2][64*KTP]
  _Float16* pbuf = (_Float16*)(uni + KVB + KTB); // [8][16*PPAD]
  float*    obuf = (float*)uni;                  // [8][16*OPAD], epilogue only

  const int t = threadIdx.x;
  const int w = t >> 5;
  const int l = t & 31;
  const int l15 = l & 15;
  const int hi = l >> 4;
  const int hi8 = hi * 8;
  const int batch = blockIdx.y;
  const int rowS = blockIdx.x * 128 + w * 16;
  const long gq0 = (long)batch * S_;
  const int rr = t >> 3;          // staging: key row 0..31
  const int colg = (t & 7) * 8;   // staging: feature col group

  // stage W_comb (64x64 f32) as f16 row-major
#pragma unroll
  for (int i = 0; i < 16; ++i) {
    int idx = t * 16 + i;
    int e = idx >> 6, f = idx & 63;
    wl[e * WPAD + f] = (_Float16)Wc[idx];
  }

  // Q A-fragments, pre-scaled by 1/sqrt(8) so scores come out scaled.
  v16h qa[2];
  {
    const _Float16* qr = q + (gq0 + rowS + l15) * E_;
#pragma unroll
    for (int c = 0; c < 2; ++c) {
      h8 lo = *(const h8*)(qr + 32 * c + hi8);
      h8 hh = *(const h8*)(qr + 32 * c + hi8 + 16);
#pragma unroll
      for (int e = 0; e < 8; ++e) {
        qa[c][e]     = (_Float16)((float)lo[e] * SCALE_);
        qa[c][8 + e] = (_Float16)((float)hh[e] * SCALE_);
      }
    }
  }

  v8f Oa[4];
#pragma unroll
  for (int n = 0; n < 4; ++n)
#pragma unroll
    for (int r = 0; r < 8; ++r) Oa[n][r] = 0.0f;
  float rowmax[8], rowsum[8];
#pragma unroll
  for (int r = 0; r < 8; ++r) { rowmax[r] = -3.0e38f; rowsum[r] = 0.0f; }

  // ---- prologue: stage tile 0 into buffer 0 ----
  {
    const _Float16* gp = q + (gq0 + 0 * 32 + rr) * E_ + colg;
    stage16_async(gp, kvb + 0 * KVE + rr * KPAD + colg);
    h8 d0 = *(const h8*)gp;
#pragma unroll
    for (int i = 0; i < 8; ++i) ktrb[0 * KTE + (colg + i) * KTP + rr] = d0[i];
  }
  wait_async_all();
  __syncthreads();

  constexpr int NT = S_ / 32;
  for (int jt = 0; jt < NT; ++jt) {
    const int cur = jt & 1;
    const int nxt = cur ^ 1;
    const bool more = (jt + 1) < NT;

    // issue next tile's async copy + register load before computing
    h8 dn = {};
    if (more) {
      const _Float16* gp = q + (gq0 + (jt + 1) * 32 + rr) * E_ + colg;
      stage16_async(gp, kvb + nxt * KVE + rr * KPAD + colg);
      dn = *(const h8*)gp;   // waitcnt sinks past the WMMA block below
    }

    const _Float16* kv  = kvb + cur * KVE;
    const _Float16* ktr = ktrb + cur * KTE;

    // ---- S = (Q/sqrt8) . K^T : two 16x16 C frags over K=64 ----
    v8f s0, s1;
#pragma unroll
    for (int r = 0; r < 8; ++r) { s0[r] = 0.0f; s1[r] = 0.0f; }
#pragma unroll
    for (int c = 0; c < 2; ++c) {
      v16h bk0, bk1;
      const _Float16* p0 = kv + l15 * KPAD + 32 * c + hi8;
      const _Float16* p1 = kv + (16 + l15) * KPAD + 32 * c + hi8;
      h8 a0 = *(const h8*)p0, a1 = *(const h8*)(p0 + 16);
      h8 c0 = *(const h8*)p1, c1 = *(const h8*)(p1 + 16);
#pragma unroll
      for (int e = 0; e < 8; ++e) {
        bk0[e] = a0[e]; bk0[8 + e] = a1[e];
        bk1[e] = c0[e]; bk1[8 + e] = c1[e];
      }
      s0 = wmma_f32_f16(qa[c], bk0, s0);
      s1 = wmma_f32_f16(qa[c], bk1, s1);
    }

    // ---- online softmax ----
#pragma unroll
    for (int r = 0; r < 8; ++r) {
      float a0 = s0[r], a1 = s1[r];
      float tmax = redmax16(fmaxf(a0, a1));
      float nmax = fmaxf(rowmax[r], tmax);
      float corr = __expf(rowmax[r] - nmax);
      rowmax[r] = nmax;
      float p0 = __expf(a0 - nmax);
      float p1 = __expf(a1 - nmax);
      rowsum[r] = rowsum[r] * corr + redsum16(p0 + p1);
#pragma unroll
      for (int n = 0; n < 4; ++n) Oa[n][r] *= corr;
      _Float16* pr = pbuf + w * PBE + (r + hi8) * PPAD;  // C layout -> LDS
      pr[l15] = (_Float16)p0;
      pr[16 + l15] = (_Float16)p1;
    }

    // P back as an A fragment (same-wave DS ops are in order)
    v16h pa;
    {
      const _Float16* pr = pbuf + w * PBE + l15 * PPAD + hi8;
      h8 lo = *(const h8*)pr;
      h8 hh = *(const h8*)(pr + 16);
#pragma unroll
      for (int e = 0; e < 8; ++e) { pa[e] = lo[e]; pa[8 + e] = hh[e]; }
    }

    // ---- O += P . V (V fragments contiguous from transposed tile) ----
#pragma unroll
    for (int n = 0; n < 4; ++n) {
      v16h vf;
      const _Float16* vr = ktr + (n * 16 + l15) * KTP + hi8;
      h8 lo = *(const h8*)vr, hh = *(const h8*)(vr + 16);
#pragma unroll
      for (int e = 0; e < 8; ++e) { vf[e] = lo[e]; vf[8 + e] = hh[e]; }
      Oa[n] = wmma_f32_f16(pa, vf, Oa[n]);
    }

    // finish next tile's transposed copy, then tile handoff
    if (more) {
#pragma unroll
      for (int i = 0; i < 8; ++i)
        ktrb[nxt * KTE + (colg + i) * KTP + rr] = dn[i];
    }
    wait_async_all();
    __syncthreads();
  }

  // ---- epilogue: normalize, re-layout via LDS, attn_out = O @ Wc^T ----
#pragma unroll
  for (int r = 0; r < 8; ++r) {
    float inv = 1.0f / rowsum[r];
    float* orow = obuf + w * OBE + (r + hi8) * OPAD;
#pragma unroll
    for (int n = 0; n < 4; ++n) orow[n * 16 + l15] = Oa[n][r] * inv;
  }
  v16h af[2];
  {
    const float* orow = obuf + w * OBE + l15 * OPAD;
#pragma unroll
    for (int c = 0; c < 2; ++c)
#pragma unroll
      for (int e = 0; e < 16; ++e) {
        int f = 32 * c + ((e & 8) << 1) + hi8 + (e & 7);
        af[c][e] = (_Float16)orow[f];
      }
  }
  v8f Cw[4];
#pragma unroll
  for (int n = 0; n < 4; ++n)
#pragma unroll
    for (int r = 0; r < 8; ++r) Cw[n][r] = 0.0f;
#pragma unroll
  for (int c = 0; c < 2; ++c)
#pragma unroll
    for (int n = 0; n < 4; ++n) {
      v16h bw;
      const _Float16* wr = wl + (n * 16 + l15) * WPAD + 32 * c + hi8;
      h8 lo = *(const h8*)wr, hh = *(const h8*)(wr + 16);
#pragma unroll
      for (int e = 0; e < 8; ++e) { bw[e] = lo[e]; bw[8 + e] = hh[e]; }
      Cw[n] = wmma_f32_f16(af[c], bw, Cw[n]);
    }

  // bias + residual, then LN1, write x1
#pragma unroll
  for (int r = 0; r < 8; ++r) {
    long grow = gq0 + rowS + r + hi8;
#pragma unroll
    for (int n = 0; n < 4; ++n) {
      int col = n * 16 + l15;
      Cw[n][r] += bcomb[col] + x[grow * E_ + col];
    }
  }
#pragma unroll
  for (int r = 0; r < 8; ++r) {
    float s = Cw[0][r] + Cw[1][r] + Cw[2][r] + Cw[3][r];
    float mean = redsum16(s) * (1.0f / 64.0f);
    float vs = 0.0f;
#pragma unroll
    for (int n = 0; n < 4; ++n) { float d = Cw[n][r] - mean; vs += d * d; }
    float rstd = rsqrtf(redsum16(vs) * (1.0f / 64.0f) + EPS_);
    long grow = gq0 + rowS + r + hi8;
#pragma unroll
    for (int n = 0; n < 4; ++n) {
      int col = n * 16 + l15;
      float d = (Cw[n][r] - mean) * rstd;
      x1[grow * E_ + col] = d * g1[col] + lnb1[col];
    }
  }
}

// ---------------------------------------------------------------------------
// Kernel 3: quantum FFN + residual + LN2.
// relu(qf @ W1^T + b1) built per-lane directly in A-fragment layout, then
// 16x16x32 WMMAs over K=256 against W2 (f16, LDS-staged).
// ---------------------------------------------------------------------------
__global__ __launch_bounds__(256) void qtb_ffn(
    const float* __restrict__ x1, const float* __restrict__ theta,
    const float* __restrict__ W1, const float* __restrict__ b1,
    const float* __restrict__ W2, const float* __restrict__ b2,
    const float* __restrict__ g2, const float* __restrict__ lnb2,
    float* __restrict__ out) {
  constexpr int W2P = 264;
  __shared__ alignas(16) float    w1l[FFN_ * NQ_];
  __shared__ float                b1l[FFN_];
  __shared__ alignas(16) _Float16 w2l[E_ * W2P];

  const int t = threadIdx.x;
  const int w = t >> 5;
  const int l = t & 31;
  const int l15 = l & 15;
  const int hi = l >> 4;
  const int hi8 = hi * 8;
  const long rowbase = (long)blockIdx.x * 128 + w * 16;

#pragma unroll
  for (int i = 0; i < 8; ++i) w1l[i * 256 + t] = W1[i * 256 + t];
  b1l[t] = b1[t];
#pragma unroll
  for (int i = 0; i < 64; ++i) {
    int idx = i * 256 + t;
    int e = idx >> 8, f = idx & 255;
    w2l[e * W2P + f] = (_Float16)W2[idx];
  }
  __syncthreads();

  float qf[8];
  {
    const float* xr = x1 + (rowbase + l15) * E_;
#pragma unroll
    for (int i = 0; i < 8; ++i) qf[i] = __cosf(xr[i]) * __cosf(theta[i]);
  }

  v8f acc[4];
#pragma unroll
  for (int n = 0; n < 4; ++n)
#pragma unroll
    for (int r = 0; r < 8; ++r) acc[n][r] = 0.0f;

#pragma unroll
  for (int c = 0; c < 8; ++c) {
    v16h a;
#pragma unroll
    for (int e = 0; e < 16; ++e) {
      int j = 32 * c + ((e & 8) << 1) + hi8 + (e & 7);  // A-layout k index
      const float* wr = w1l + j * 8;
      float h = b1l[j];
#pragma unroll
      for (int i = 0; i < 8; ++i) h = fmaf(qf[i], wr[i], h);
      a[e] = (_Float16)fmaxf(h, 0.0f);
    }
#pragma unroll
    for (int n = 0; n < 4; ++n) {
      v16h bw;
      const _Float16* wr2 = w2l + (n * 16 + l15) * W2P + 32 * c + hi8;
      h8 lo = *(const h8*)wr2, hh = *(const h8*)(wr2 + 16);
#pragma unroll
      for (int e = 0; e < 8; ++e) { bw[e] = lo[e]; bw[8 + e] = hh[e]; }
      acc[n] = wmma_f32_f16(a, bw, acc[n]);
    }
  }

  // bias + residual + LN2 + store
#pragma unroll
  for (int r = 0; r < 8; ++r) {
    long grow = rowbase + r + hi8;
#pragma unroll
    for (int n = 0; n < 4; ++n) {
      int col = n * 16 + l15;
      acc[n][r] += b2[col] + x1[grow * E_ + col];
    }
  }
#pragma unroll
  for (int r = 0; r < 8; ++r) {
    float s = acc[0][r] + acc[1][r] + acc[2][r] + acc[3][r];
    float mean = redsum16(s) * (1.0f / 64.0f);
    float vs = 0.0f;
#pragma unroll
    for (int n = 0; n < 4; ++n) { float d = acc[n][r] - mean; vs += d * d; }
    float rstd = rsqrtf(redsum16(vs) * (1.0f / 64.0f) + EPS_);
    long grow = rowbase + r + hi8;
#pragma unroll
    for (int n = 0; n < 4; ++n) {
      int col = n * 16 + l15;
      float d = (acc[n][r] - mean) * rstd;
      out[grow * E_ + col] = d * g2[col] + lnb2[col];
    }
  }
}

// ---------------------------------------------------------------------------
extern "C" void kernel_launch(void* const* d_in, const int* in_sizes, int n_in,
                              void* d_out, int out_size, void* d_ws,
                              size_t ws_size, hipStream_t stream) {
  const float* x          = (const float*)d_in[0];
  const float* theta_attn = (const float*)d_in[1];
  const float* W_comb     = (const float*)d_in[2];
  const float* b_comb     = (const float*)d_in[3];
  const float* ln1_g      = (const float*)d_in[4];
  const float* ln1_b      = (const float*)d_in[5];
  const float* theta_ffn  = (const float*)d_in[6];
  const float* W1         = (const float*)d_in[7];
  const float* b1         = (const float*)d_in[8];
  const float* W2         = (const float*)d_in[9];
  const float* b2         = (const float*)d_in[10];
  const float* ln2_g      = (const float*)d_in[11];
  const float* ln2_b      = (const float*)d_in[12];
  float* out = (float*)d_out;

  // workspace: q (f16, 2 MB) | x1 (f32, 4 MB)
  _Float16* qbuf = (_Float16*)d_ws;
  float* x1buf = (float*)((char*)d_ws + (size_t)ROWS_ * E_ * sizeof(_Float16));

  qtb_qfeat<<<(ROWS_ * H_) / 256, 256, 0, stream>>>(x, theta_attn, qbuf);

  dim3 g2(S_ / 128, B_);
  qtb_attn<<<g2, 256, 0, stream>>>(qbuf, x, W_comb, b_comb, ln1_g, ln1_b,
                                   x1buf);

  qtb_ffn<<<ROWS_ / 128, 256, 0, stream>>>(x1buf, theta_ffn, W1, b1, W2, b2,
                                           ln2_g, ln2_b, out);
}
